// DynamicEdgeConv_81664508166585
// MI455X (gfx1250) — matrix operand order repaired and checked by reading, not compiled, not run on hardware
//
#include <hip/hip_runtime.h>
#include <hip/hip_bf16.h>

typedef __bf16 v16bf __attribute__((ext_vector_type(16)));
typedef float  v8f   __attribute__((ext_vector_type(8)));
typedef unsigned int u32x4 __attribute__((ext_vector_type(4)));

#define D_IN   128
#define EDIM   256
#define KNN    16
#define NPTS   2048
#define PPB    4            // points per block -> M = 64 rows
#define MROWS  (KNN * PPB)  // 64
#define LN_EPS 1e-5f

// dynamic LDS partition (bytes)
#define OFF_HF      0                              // 64*256 f32   = 65536
#define OFF_HB      (OFF_HF + MROWS * EDIM * 4)    // 64*256 bf16  = 32768
#define OFF_PSUM    (OFF_HB + MROWS * EDIM * 2)    // 256 f32      = 1024
#define OFF_PSUM2   (OFF_PSUM + 256 * 4)           // 256 f32      = 1024
#define OFF_CENTRAL (OFF_PSUM2 + 256 * 4)          // 4*128 f32    = 2048
#define OFF_SMU     (OFF_CENTRAL + PPB * D_IN * 4) // 64 f32
#define OFF_SRV     (OFF_SMU + MROWS * 4)          // 64 f32
#define SHBYTES     (OFF_SRV + MROWS * 4)          // ~102.5 KB (WGP has 320 KB)

__device__ __forceinline__ unsigned short f2bf(float f) {
  unsigned u = __float_as_uint(f);
  u += 0x7FFFu + ((u >> 16) & 1u);   // round-to-nearest-even
  return (unsigned short)(u >> 16);
}

__device__ __forceinline__ float gelu_exact(float x) {
  return 0.5f * x * (1.0f + erff(x * 0.70710678118654752440f));
}

union FragBF { v16bf v; u32x4 q[2]; };

__device__ __forceinline__ v8f wmma_bf16(const FragBF& a, const FragBF& b, v8f c) {
  return __builtin_amdgcn_wmma_f32_16x16x32_bf16(false, a.v, false, b.v,
                                                 (short)0, c, false, false);
}

// ---------------------------------------------------------------------------
// Pre-pass: f32 weights -> bf16, swizzled into WMMA B-fragment order.
// (kt, nt, lane, i) <- W[kt*32 + (lane>=16?16:0) + i][nt*16 + (lane&15)]
// ---------------------------------------------------------------------------
__global__ void swizzle_weights_kernel(const float* __restrict__ W1,
                                       const float* __restrict__ W2,
                                       const float* __restrict__ W3,
                                       unsigned short* __restrict__ o) {
  int id    = blockIdx.x * blockDim.x + threadIdx.x;   // 3 * 65536 total
  int layer = id >> 16;
  int r     = id & 0xFFFF;
  int i     = r & 15;
  int lane  = (r >> 4) & 31;
  int nt    = (r >> 9) & 15;
  int kt    = (r >> 13) & 7;
  int row   = kt * 32 + ((lane >> 4) & 1) * 16 + i;
  int col   = nt * 16 + (lane & 15);
  const float* W = (layer == 0) ? W1 : (layer == 1) ? W2 : W3;
  o[(size_t)layer * 65536 + r] = f2bf(W[row * EDIM + col]);
}

// ---------------------------------------------------------------------------
// kNN: one thread per query, LDS-tiled candidates, register top-16.
// ---------------------------------------------------------------------------
__global__ void __launch_bounds__(256)
knn_kernel(const float* __restrict__ coords, int* __restrict__ knn_out) {
  __shared__ __align__(16) float sc[256 * 3];
  const int tid = threadIdx.x;
  const int b   = blockIdx.x / (NPTS / 256);
  const int n   = (blockIdx.x % (NPTS / 256)) * 256 + tid;
  const float* cb = coords + (size_t)b * NPTS * 3;
  const float cx = cb[n * 3 + 0], cy = cb[n * 3 + 1], cz = cb[n * 3 + 2];

  float bd[KNN]; int bi[KNN];
  #pragma unroll
  for (int j = 0; j < KNN; ++j) { bd[j] = 3.4e38f; bi[j] = 0; }

  for (int t = 0; t < NPTS; t += 256) {
    __syncthreads();
    sc[tid * 3 + 0] = cb[(t + tid) * 3 + 0];
    sc[tid * 3 + 1] = cb[(t + tid) * 3 + 1];
    sc[tid * 3 + 2] = cb[(t + tid) * 3 + 2];
    __syncthreads();
    #pragma unroll 1
    for (int m = 0; m < 256; ++m) {
      float dx = cx - sc[m * 3 + 0];
      float dy = cy - sc[m * 3 + 1];
      float dz = cz - sc[m * 3 + 2];
      float d  = fmaxf(dx * dx + dy * dy + dz * dz, 0.0f);
      if (d < bd[KNN - 1]) {
        bd[KNN - 1] = d; bi[KNN - 1] = t + m;
        #pragma unroll
        for (int j = KNN - 1; j > 0; --j) {
          if (bd[j] < bd[j - 1]) {
            float td = bd[j]; bd[j] = bd[j - 1]; bd[j - 1] = td;
            int   ti = bi[j]; bi[j] = bi[j - 1]; bi[j - 1] = ti;
          }
        }
      }
    }
  }
  int* o = knn_out + ((size_t)b * NPTS + n) * KNN;
  #pragma unroll
  for (int j = 0; j < KNN; ++j) o[j] = bi[j];
}

// ---------------------------------------------------------------------------
// One (Linear -> LN -> GELU) layer for the 64x256 tile in LDS.
// Per wave: 2 N-tiles x 4 M-tiles x 8 K-steps = 64 wmma; each B fragment
// is reused across 4 M-tiles (4x less L2 weight traffic per point).
// ---------------------------------------------------------------------------
template<bool LAST>
__device__ __forceinline__
void mlp_layer(const unsigned short* __restrict__ wsw,
               const unsigned short* __restrict__ wnext,   // prefetch target (may be null)
               const float* __restrict__ bias,
               const float* __restrict__ g,
               const float* __restrict__ be,
               unsigned short* hb, float* hf,
               float* psum, float* psum2, float* smu, float* srv) {
  const int tid  = threadIdx.x;
  const int lane = tid & 31, wave = tid >> 5;
  const int row  = lane & 15, hs = (lane >> 4) & 1;
  const int nt0  = wave * 2, nt1 = nt0 + 1;

  v8f acc[PPB][2];
  #pragma unroll
  for (int mt = 0; mt < PPB; ++mt) {
    acc[mt][0] = (v8f){0,0,0,0,0,0,0,0};
    acc[mt][1] = (v8f){0,0,0,0,0,0,0,0};
  }

  #pragma unroll
  for (int kt = 0; kt < 8; ++kt) {
    const int off0 = (((kt * 16 + nt0) * 32) + lane) * 16;
    const int off1 = (((kt * 16 + nt1) * 32) + lane) * 16;
    FragBF b0, b1;
    const u32x4* bp0 = (const u32x4*)&wsw[off0];
    b0.q[0] = bp0[0]; b0.q[1] = bp0[1];
    const u32x4* bp1 = (const u32x4*)&wsw[off1];
    b1.q[0] = bp1[0]; b1.q[1] = bp1[1];
    if (wnext) {  // warm WGP$ with next layer's fragments (global_prefetch_b8)
      __builtin_prefetch((const void*)&wnext[off0], 0, 1);
      __builtin_prefetch((const void*)&wnext[off1], 0, 1);
    }
    #pragma unroll
    for (int mt = 0; mt < PPB; ++mt) {
      FragBF af;
      const u32x4* ap = (const u32x4*)&hb[(mt * 16 + row) * EDIM + kt * 32 + hs * 16];
      af.q[0] = ap[0];   // K = base + 0..7
      af.q[1] = ap[1];   // K = base + 8..15
      acc[mt][0] = wmma_bf16(af, b0, acc[mt][0]);
      acc[mt][1] = wmma_bf16(af, b1, acc[mt][1]);
    }
  }

  // scatter D + bias (C/D layout: VGPR j -> row j + hs*8, col = lane&15)
  {
    const int c0 = nt0 * 16 + (lane & 15);
    const int c1 = nt1 * 16 + (lane & 15);
    const float bb0 = bias[c0], bb1 = bias[c1];
    #pragma unroll
    for (int mt = 0; mt < PPB; ++mt) {
      #pragma unroll
      for (int j = 0; j < 8; ++j) {
        const int rr = mt * 16 + j + hs * 8;
        hf[rr * EDIM + c0] = acc[mt][0][j] + bb0;
        hf[rr * EDIM + c1] = acc[mt][1][j] + bb1;
      }
    }
  }
  __syncthreads();

  // LN stats: 4 threads per row (64 rows), then combine
  {
    const int r = tid >> 2, q = tid & 3;
    float s = 0.0f, s2 = 0.0f;
    #pragma unroll 4
    for (int c = q * 64; c < q * 64 + 64; ++c) {
      float v = hf[r * EDIM + c];
      s += v; s2 += v * v;
    }
    psum[tid] = s; psum2[tid] = s2;
  }
  __syncthreads();
  if (tid < MROWS) {
    float s  = psum[tid * 4] + psum[tid * 4 + 1] + psum[tid * 4 + 2] + psum[tid * 4 + 3];
    float s2 = psum2[tid * 4] + psum2[tid * 4 + 1] + psum2[tid * 4 + 2] + psum2[tid * 4 + 3];
    const float mu  = s * (1.0f / EDIM);
    const float var = s2 * (1.0f / EDIM) - mu * mu;
    smu[tid] = mu;
    srv[tid] = rsqrtf(var + LN_EPS);
  }
  __syncthreads();

  // apply LN + exact GELU; bf16 for next layer, f32 kept if last
  {
    const int c0 = (tid & 15) * 16;
    #pragma unroll
    for (int p = 0; p < PPB; ++p) {
      const int r = p * 16 + (tid >> 4);
      const float mu = smu[r], rv = srv[r];
      #pragma unroll
      for (int e = 0; e < 16; ++e) {
        const int c = c0 + e;
        const float v   = (hf[r * EDIM + c] - mu) * rv * g[c] + be[c];
        const float act = gelu_exact(v);
        if (LAST) hf[r * EDIM + c] = act;
        else      hb[r * EDIM + c] = f2bf(act);
      }
    }
  }
  __syncthreads();
}

// ---------------------------------------------------------------------------
// Fused kernel: 4 points per block -> edge [64,256] bf16 -> 3 WMMA layers ->
// max over K per point -> + shortcut -> GELU -> out.
// ---------------------------------------------------------------------------
__global__ void __launch_bounds__(256)
edgeconv_kernel(const float* __restrict__ x,
                const int* __restrict__ knn_idx,
                const unsigned short* __restrict__ w1,
                const unsigned short* __restrict__ w2,
                const unsigned short* __restrict__ w3,
                const float* __restrict__ b1, const float* __restrict__ g1, const float* __restrict__ be1,
                const float* __restrict__ b2, const float* __restrict__ g2, const float* __restrict__ be2,
                const float* __restrict__ b3, const float* __restrict__ g3, const float* __restrict__ be3,
                const float* __restrict__ Ws, const float* __restrict__ bs,
                float* __restrict__ out) {
  extern __shared__ __align__(16) char smem[];
  float*          hf      = (float*)(smem + OFF_HF);
  unsigned short* hb      = (unsigned short*)(smem + OFF_HB);
  float*          psum    = (float*)(smem + OFF_PSUM);
  float*          psum2   = (float*)(smem + OFF_PSUM2);
  float*          central = (float*)(smem + OFF_CENTRAL);
  float*          smu     = (float*)(smem + OFF_SMU);
  float*          srv     = (float*)(smem + OFF_SRV);

  const int tid = threadIdx.x;
  const int pt0 = blockIdx.x * PPB;      // 4 consecutive points, same batch

  // build edge features for 4 points: [central | neighbor - central] -> bf16
  {
    const int rloc = tid >> 4;
    const int c0   = (tid & 15) * 16;
    #pragma unroll
    for (int p = 0; p < PPB; ++p) {
      const int pt = pt0 + p;
      const int b  = pt / NPTS;
      const int r  = p * 16 + rloc;
      const int nb = knn_idx[(size_t)pt * KNN + rloc];
      const float* xr = x + (size_t)pt * D_IN;
      const float* xn = x + ((size_t)b * NPTS + nb) * D_IN;
      if (c0 < D_IN) {
        #pragma unroll
        for (int e = 0; e < 16; ++e) {
          const float v = xr[c0 + e];
          hb[r * EDIM + c0 + e] = f2bf(v);
          if (rloc == 0) central[p * D_IN + c0 + e] = v;
        }
      } else {
        #pragma unroll
        for (int e = 0; e < 16; ++e) {
          const int d = c0 + e - D_IN;
          hb[r * EDIM + c0 + e] = f2bf(xn[d] - xr[d]);
        }
      }
    }
  }
  __syncthreads();

  mlp_layer<false>(w1, w2,      b1, g1, be1, hb, hf, psum, psum2, smu, srv);
  mlp_layer<false>(w2, w3,      b2, g2, be2, hb, hf, psum, psum2, smu, srv);
  mlp_layer<true >(w3, nullptr, b3, g3, be3, hb, hf, psum, psum2, smu, srv);

  // shortcut GEMV for 4 points; each Ws element loaded once, reused 4x
  float s0 = bs[tid], s1 = s0, s2 = s0, s3 = s0;
  #pragma unroll 4
  for (int d = 0; d < D_IN; ++d) {
    const float w = Ws[d * EDIM + tid];
    s0 += central[0 * D_IN + d] * w;
    s1 += central[1 * D_IN + d] * w;
    s2 += central[2 * D_IN + d] * w;
    s3 += central[3 * D_IN + d] * w;
  }
  const float sc[PPB] = {s0, s1, s2, s3};

  #pragma unroll
  for (int p = 0; p < PPB; ++p) {
    float agg = -3.4e38f;
    #pragma unroll
    for (int r = 0; r < KNN; ++r) agg = fmaxf(agg, hf[(p * 16 + r) * EDIM + tid]);
    out[(size_t)(pt0 + p) * EDIM + tid] = gelu_exact(agg + sc[p]);
  }
}

// ---------------------------------------------------------------------------
extern "C" void kernel_launch(void* const* d_in, const int* in_sizes, int n_in,
                              void* d_out, int out_size, void* d_ws, size_t ws_size,
                              hipStream_t stream) {
  const float* x   = (const float*)d_in[0];
  const float* co  = (const float*)d_in[1];
  const float* W1  = (const float*)d_in[2];
  const float* b1  = (const float*)d_in[3];
  const float* g1  = (const float*)d_in[4];
  const float* be1 = (const float*)d_in[5];
  const float* W2  = (const float*)d_in[6];
  const float* b2  = (const float*)d_in[7];
  const float* g2  = (const float*)d_in[8];
  const float* be2 = (const float*)d_in[9];
  const float* W3  = (const float*)d_in[10];
  const float* b3  = (const float*)d_in[11];
  const float* g3  = (const float*)d_in[12];
  const float* be3 = (const float*)d_in[13];
  const float* Ws  = (const float*)d_in[14];
  const float* bs  = (const float*)d_in[15];

  const int B    = in_sizes[0] / (NPTS * D_IN);
  const int Ntot = B * NPTS;

  // workspace: [3 x 65536 bf16 swizzled weights][B*N*16 int knn indices]
  unsigned short* wsw = (unsigned short*)d_ws;
  int* knn = (int*)((char*)d_ws + (size_t)3 * 65536 * sizeof(unsigned short));

  swizzle_weights_kernel<<<(3 * 65536) / 256, 256, 0, stream>>>(W1, W2, W3, wsw);
  knn_kernel<<<Ntot / 256, 256, 0, stream>>>(co, knn);
  edgeconv_kernel<<<Ntot / PPB, 256, SHBYTES, stream>>>(
      x, knn, wsw, wsw + 65536, wsw + 2 * 65536,
      b1, g1, be1, b2, g2, be2, b3, g3, be3, Ws, bs, (float*)d_out);
}
